// MultiHeadAttention_88897233093239
// MI455X (gfx1250) — matrix operand order repaired and checked by reading, not compiled, not run on hardware
//
#include <hip/hip_runtime.h>
#include <hip/hip_bf16.h>

// ---- problem constants (reference: B=2, S=2048, D=1024, H=16, HD=64) ----
#define BB 2
#define SS 2048
#define DD 1024
#define HH 16
#define HDIM 64

typedef __attribute__((ext_vector_type(16))) __bf16 v16bf;
typedef __attribute__((ext_vector_type(8)))  __bf16 v8bf;
typedef __attribute__((ext_vector_type(4)))  __bf16 v4bf;
typedef __attribute__((ext_vector_type(8)))  float  v8f;

// ---------------------------------------------------------------------------
// WMMA fragment loaders (wave32, 16x16x32 bf16)
//
// A fragment (16 rows x 32 K), contiguous k order (same as load_bfrag):
//   lanes 0-15 : row = lane,    elems 0-15 -> K k0..k0+15
//   lanes16-31 : row = lane-16, elems 0-15 -> K k0+16..k0+31
// => two adjacent 16-byte loads per lane from row-major [rows, K] data.
static __device__ __forceinline__ v16bf load_afrag(const __bf16* __restrict__ A,
                                                   int lda, int m0, int k0, int lane) {
  int row = m0 + (lane & 15);
  int kb  = k0 + ((lane >> 4) << 4);
  const __bf16* p = A + (size_t)row * lda + kb;
  v8bf lo = *(const v8bf*)(p);
  v8bf hi = *(const v8bf*)(p + 8);
  return __builtin_shufflevector(lo, hi, 0,1,2,3,4,5,6,7,8,9,10,11,12,13,14,15);
}

// B fragment (32 K x 16 cols): lane n holds col n, K k0..k0+15 (lanes 0-15)
// or K k0+16..k0+31 (lanes 16-31) => one contiguous 32-byte load per lane
// from row-major [cols, K] data (i.e. W[N,K] gives B = W^T for free).
static __device__ __forceinline__ v16bf load_bfrag(const __bf16* __restrict__ Bm,
                                                   int ldb, int n0, int k0, int lane) {
  int col = n0 + (lane & 15);
  int kb  = k0 + ((lane >> 4) << 4);
  return *(const v16bf*)(Bm + (size_t)col * ldb + kb);
}

static __device__ __forceinline__ v8f wmma_bf16(v16bf a, v16bf b, v8f c) {
  return __builtin_amdgcn_wmma_f32_16x16x32_bf16(false, a, false, b, (short)0, c,
                                                 false, false);
}

// ---------------------------------------------------------------------------
// fp32 -> bf16 conversion (vectorized x4)
__global__ __launch_bounds__(256) void cvt_bf16_kernel(const float* __restrict__ s,
                                                       __bf16* __restrict__ d, int n4) {
  int i = blockIdx.x * blockDim.x + threadIdx.x;
  if (i < n4) {
    float4 f = ((const float4*)s)[i];
    v4bf o;
    o.x = (__bf16)f.x; o.y = (__bf16)f.y; o.z = (__bf16)f.z; o.w = (__bf16)f.w;
    ((v4bf*)d)[i] = o;
  }
}

// ---------------------------------------------------------------------------
// QKV projection: qkv = x @ w_qkv^T + b_qkv, split into per-head buffers:
//   Q (pre-scaled by 1/sqrt(HD)) -> [B,H,S,HD], K -> [B,H,S,HD], V -> [B,H,HD,S]
// One wave per 32x32 output tile (2x2 WMMA register tiling: 2 A-frags + 2
// B-frags feed 4 WMMAs per k-step => 2x flops/byte vs 1 tile/wave).
static __device__ __forceinline__ void qkv_store16(
    const v8f& acc, int m0, int n0, int lane,
    const float* __restrict__ bias,
    __bf16* __restrict__ qh, __bf16* __restrict__ kh, __bf16* __restrict__ vth) {
  // C/D layout: lane l holds col n0+(l&15); VGPR j holds row m0 + 8*(l>>4) + j
  int n   = n0 + (lane & 15);
  float bz = bias[n];
  int sec = n >> 10;                    // 0=Q 1=K 2=V
  int nn  = n & (DD - 1);
  int hh  = nn >> 6;
  int dd  = nn & (HDIM - 1);
  int rbase = m0 + ((lane >> 4) << 3);
#pragma unroll
  for (int j = 0; j < 8; ++j) {
    int r  = rbase + j;
    int bb = r >> 11;                   // / S
    int s  = r & (SS - 1);
    float v = acc[j] + bz;
    int bhs = bb * HH + hh;
    if (sec == 0)
      qh[((size_t)(bhs * SS + s)) * HDIM + dd] = (__bf16)(v * 0.125f); // 1/sqrt(64)
    else if (sec == 1)
      kh[((size_t)(bhs * SS + s)) * HDIM + dd] = (__bf16)v;
    else
      vth[((size_t)(bhs * HDIM + dd)) * SS + s] = (__bf16)v;           // transposed
  }
}

__global__ __launch_bounds__(256) void qkv_gemm_kernel(
    const __bf16* __restrict__ A,     // [B*S, D]
    const __bf16* __restrict__ W,     // [3D, D]
    const float*  __restrict__ bias,  // [3D]
    __bf16* __restrict__ qh, __bf16* __restrict__ kh, __bf16* __restrict__ vth) {
  int wid  = (blockIdx.x * blockDim.x + threadIdx.x) >> 5;
  int lane = threadIdx.x & 31;
  const int NT = (3 * DD) / 32;          // 96 tiles along N (32-wide)
  int mt = wid / NT, nt = wid % NT;
  int m0 = mt * 32, n0 = nt * 32;

  v8f a00 = {}, a01 = {}, a10 = {}, a11 = {};
  for (int k0 = 0; k0 < DD; k0 += 32) {
    v16bf af0 = load_afrag(A, DD, m0,      k0, lane);
    v16bf af1 = load_afrag(A, DD, m0 + 16, k0, lane);
    v16bf bf0 = load_bfrag(W, DD, n0,      k0, lane);
    v16bf bf1 = load_bfrag(W, DD, n0 + 16, k0, lane);
    a00 = wmma_bf16(af0, bf0, a00);
    a01 = wmma_bf16(af0, bf1, a01);
    a10 = wmma_bf16(af1, bf0, a10);
    a11 = wmma_bf16(af1, bf1, a11);
  }
  qkv_store16(a00, m0,      n0,      lane, bias, qh, kh, vth);
  qkv_store16(a01, m0,      n0 + 16, lane, bias, qh, kh, vth);
  qkv_store16(a10, m0 + 16, n0,      lane, bias, qh, kh, vth);
  qkv_store16(a11, m0 + 16, n0 + 16, lane, bias, qh, kh, vth);
}

// ---------------------------------------------------------------------------
// Flash attention, transposed-score formulation. One wave per (b,h, 16-query tile).
//   S^T = K * Q^T      (A = K rows, B = Q rows as B-frag)
//   O^T = V^T * P^T    (A = V^T rows, B = P^T built from score frags + shfl_xor(16))
__global__ __launch_bounds__(256) void attn_kernel(
    const __bf16* __restrict__ qh, const __bf16* __restrict__ kh,
    const __bf16* __restrict__ vth, __bf16* __restrict__ attn) {
  int wid  = (blockIdx.x * blockDim.x + threadIdx.x) >> 5;
  int lane = threadIdx.x & 31;
  const int QT = SS / 16;               // 128 query tiles per (b,h)
  int bh = wid / QT, qt = wid % QT;     // 8 waves of a block share (b,h) -> WGP$ reuse
  int q0 = qt * 16;
  int h  = bh % HH, b = bh / HH;

  const __bf16* Q  = qh  + (size_t)bh * SS * HDIM;
  const __bf16* Kp = kh  + (size_t)bh * SS * HDIM;
  const __bf16* Vt = vth + (size_t)bh * HDIM * SS;
  float slope = exp2f(-0.5f * (float)(h + 1));   // ALiBi: exp2(-(h+1)*8/H)

  // Q B-fragments are loop-invariant (HD=64 -> two K-chunks of 32)
  v16bf qb0 = load_bfrag(Q, HDIM, q0, 0,  lane);
  v16bf qb1 = load_bfrag(Q, HDIM, q0, 32, lane);

  int  qcol  = q0 + (lane & 15);        // this lane's query index
  int  koff  = (lane >> 4) << 3;        // 0 or 8: key-row offset inside subtile
  bool upper = (lane >= 16);

  float m_run = -1e30f, l_run = 0.f;
  v8f ot0 = {}, ot1 = {}, ot2 = {}, ot3 = {};

  for (int kb0 = 0; kb0 < q0 + 16; kb0 += 32) {   // causal bound
    // scores^T for 32 keys x 16 queries (two 16-key subtiles)
    v8f sta = {}, stb = {};
    sta = wmma_bf16(load_afrag(Kp, HDIM, kb0,      0,  lane), qb0, sta);
    sta = wmma_bf16(load_afrag(Kp, HDIM, kb0,      32, lane), qb1, sta);
    stb = wmma_bf16(load_afrag(Kp, HDIM, kb0 + 16, 0,  lane), qb0, stb);
    stb = wmma_bf16(load_afrag(Kp, HDIM, kb0 + 16, 32, lane), qb1, stb);

    // ALiBi bias + causal mask + local max (per lane = per query column)
    float pa[8], pb[8];
    float mloc = -1e30f;
#pragma unroll
    for (int j = 0; j < 8; ++j) {
      int ka = kb0 + koff + j;
      int kc = ka + 16;
      float sa = (ka <= qcol) ? (sta[j] + slope * (float)(ka - qcol)) : -1e30f;
      float sb = (kc <= qcol) ? (stb[j] + slope * (float)(kc - qcol)) : -1e30f;
      pa[j] = sa; pb[j] = sb;
      mloc = fmaxf(mloc, fmaxf(sa, sb));
    }
    mloc = fmaxf(mloc, __shfl_xor(mloc, 16, 32));  // combine lane-halves of a query
    float mnew = fmaxf(m_run, mloc);
    float corr = __expf(m_run - mnew);
    m_run = mnew;

    float rsum = 0.f;
#pragma unroll
    for (int j = 0; j < 8; ++j) {
      pa[j] = __expf(pa[j] - mnew);
      pb[j] = __expf(pb[j] - mnew);
      rsum += pa[j] + pb[j];
    }
    l_run = l_run * corr + rsum + __shfl_xor(rsum, 16, 32);

    // Build P^T B-fragment: lane n needs keys 0..15 (lanes<16) / 16..31 (lanes>=16)
    // of this 32-key block; half of them live in lane^16 -> one shfl_xor each.
    v16bf pf;
#pragma unroll
    for (int j = 0; j < 8; ++j) {
      float ax = __shfl_xor(pa[j], 16, 32);
      float bx = __shfl_xor(pb[j], 16, 32);
      pf[j]     = (__bf16)(upper ? bx    : pa[j]);
      pf[j + 8] = (__bf16)(upper ? pb[j] : ax);
    }

    // rescale running O^T and accumulate O^T += V^T(dims x keys) * P^T
#pragma unroll
    for (int j = 0; j < 8; ++j) {
      ot0[j] *= corr; ot1[j] *= corr; ot2[j] *= corr; ot3[j] *= corr;
    }
    ot0 = wmma_bf16(load_afrag(Vt, SS, 0,  kb0, lane), pf, ot0);
    ot1 = wmma_bf16(load_afrag(Vt, SS, 16, kb0, lane), pf, ot1);
    ot2 = wmma_bf16(load_afrag(Vt, SS, 32, kb0, lane), pf, ot2);
    ot3 = wmma_bf16(load_afrag(Vt, SS, 48, kb0, lane), pf, ot3);
  }

  // normalize and store merged-head layout attn[(b*S + q)*D + h*HD + dim]
  float inv = 1.0f / l_run;
  size_t rowbase = ((size_t)(b * SS + qcol)) * DD + h * HDIM + koff;
  v8bf o;
#pragma unroll
  for (int j = 0; j < 8; ++j) o[j] = (__bf16)(ot0[j] * inv);
  *(v8bf*)(attn + rowbase + 0)  = o;
#pragma unroll
  for (int j = 0; j < 8; ++j) o[j] = (__bf16)(ot1[j] * inv);
  *(v8bf*)(attn + rowbase + 16) = o;
#pragma unroll
  for (int j = 0; j < 8; ++j) o[j] = (__bf16)(ot2[j] * inv);
  *(v8bf*)(attn + rowbase + 32) = o;
#pragma unroll
  for (int j = 0; j < 8; ++j) o[j] = (__bf16)(ot3[j] * inv);
  *(v8bf*)(attn + rowbase + 48) = o;
}

// ---------------------------------------------------------------------------
// Output projection: out = attn @ w_out^T + b_out (fp32 output), 32x32 per wave.
static __device__ __forceinline__ void out_store16(
    const v8f& acc, int m0, int n0, int lane,
    const float* __restrict__ bias, float* __restrict__ out) {
  int n = n0 + (lane & 15);
  float bz = bias[n];
  int rbase = m0 + ((lane >> 4) << 3);
#pragma unroll
  for (int j = 0; j < 8; ++j)
    out[(size_t)(rbase + j) * DD + n] = acc[j] + bz;
}

__global__ __launch_bounds__(256) void out_gemm_kernel(
    const __bf16* __restrict__ A,     // [B*S, D]
    const __bf16* __restrict__ W,     // [D, D]
    const float*  __restrict__ bias,  // [D]
    float* __restrict__ out) {
  int wid  = (blockIdx.x * blockDim.x + threadIdx.x) >> 5;
  int lane = threadIdx.x & 31;
  const int NT = DD / 32;              // 32 tiles along N (32-wide)
  int mt = wid / NT, nt = wid % NT;
  int m0 = mt * 32, n0 = nt * 32;

  v8f a00 = {}, a01 = {}, a10 = {}, a11 = {};
  for (int k0 = 0; k0 < DD; k0 += 32) {
    v16bf af0 = load_afrag(A, DD, m0,      k0, lane);
    v16bf af1 = load_afrag(A, DD, m0 + 16, k0, lane);
    v16bf bf0 = load_bfrag(W, DD, n0,      k0, lane);
    v16bf bf1 = load_bfrag(W, DD, n0 + 16, k0, lane);
    a00 = wmma_bf16(af0, bf0, a00);
    a01 = wmma_bf16(af0, bf1, a01);
    a10 = wmma_bf16(af1, bf0, a10);
    a11 = wmma_bf16(af1, bf1, a11);
  }
  out_store16(a00, m0,      n0,      lane, bias, out);
  out_store16(a01, m0,      n0 + 16, lane, bias, out);
  out_store16(a10, m0 + 16, n0,      lane, bias, out);
  out_store16(a11, m0 + 16, n0 + 16, lane, bias, out);
}

// ---------------------------------------------------------------------------
extern "C" void kernel_launch(void* const* d_in, const int* in_sizes, int n_in,
                              void* d_out, int out_size, void* d_ws, size_t ws_size,
                              hipStream_t stream) {
  (void)in_sizes; (void)n_in; (void)out_size; (void)ws_size;
  const float* x     = (const float*)d_in[0];
  const float* w_qkv = (const float*)d_in[1];
  const float* b_qkv = (const float*)d_in[2];
  const float* w_out = (const float*)d_in[3];
  const float* b_out = (const float*)d_in[4];
  // d_in[5] = block_mask: causal tril, applied analytically in-kernel
  float* out = (float*)d_out;

  char* ws = (char*)d_ws;
  size_t off = 0;
  auto alloc = [&](size_t bytes) -> char* {
    char* p = ws + off;
    off = (off + bytes + 255) & ~(size_t)255;
    return p;
  };
  __bf16* xh    = (__bf16*)alloc((size_t)BB * SS * DD * 2);       // 8 MB
  __bf16* wqkvh = (__bf16*)alloc((size_t)3 * DD * DD * 2);        // 6 MB
  __bf16* wouth = (__bf16*)alloc((size_t)DD * DD * 2);            // 2 MB
  __bf16* qh    = (__bf16*)alloc((size_t)BB * HH * SS * HDIM * 2);// 8 MB
  __bf16* kh    = (__bf16*)alloc((size_t)BB * HH * SS * HDIM * 2);// 8 MB
  __bf16* vth   = (__bf16*)alloc((size_t)BB * HH * HDIM * SS * 2);// 8 MB
  __bf16* attnh = (__bf16*)alloc((size_t)BB * SS * DD * 2);       // 8 MB

  const int nx  = BB * SS * DD;    // 4194304
  const int nw1 = 3 * DD * DD;     // 3145728
  const int nw2 = DD * DD;         // 1048576
  cvt_bf16_kernel<<<(nx  / 4 + 255) / 256, 256, 0, stream>>>(x,     xh,    nx  / 4);
  cvt_bf16_kernel<<<(nw1 / 4 + 255) / 256, 256, 0, stream>>>(w_qkv, wqkvh, nw1 / 4);
  cvt_bf16_kernel<<<(nw2 / 4 + 255) / 256, 256, 0, stream>>>(w_out, wouth, nw2 / 4);

  const int qkv_waves = (BB * SS / 32) * (3 * DD / 32);  // 128*96 = 12288
  qkv_gemm_kernel<<<qkv_waves / 8, 256, 0, stream>>>(xh, wqkvh, b_qkv, qh, kh, vth);

  const int attn_waves = BB * HH * (SS / 16);            // 4096
  attn_kernel<<<attn_waves / 8, 256, 0, stream>>>(qh, kh, vth, attnh);

  const int out_waves = (BB * SS / 32) * (DD / 32);      // 128*32 = 4096
  out_gemm_kernel<<<out_waves / 8, 256, 0, stream>>>(attnh, wouth, b_out, out);
}
